// Wavenet_1554778161816
// MI455X (gfx1250) — matrix operand order repaired and checked by reading, not compiled, not run on hardware
//
#include <hip/hip_runtime.h>
#include <hip/hip_bf16.h>

// ---------------------------------------------------------------------------
// WaveNet forward for MI455X (gfx1250), wave32 + v_wmma_f32_16x16x32_f16.
// Activations stored time-major [B][T][C] in f16 so a WMMA B-fragment
// (16 consecutive K-channels for one timestep column) is one contiguous
// 32-byte load. f32 accumulation via WMMA. x tiles are staged into LDS via
// the CDNA5 async Global->LDS path (ASYNCcnt) when the builtins exist.
// ---------------------------------------------------------------------------

typedef _Float16 v16h __attribute__((ext_vector_type(16)));
typedef _Float16 h8   __attribute__((ext_vector_type(8)));
typedef _Float16 h2t  __attribute__((ext_vector_type(2)));
typedef float    v8f  __attribute__((ext_vector_type(8)));
typedef int      v4i  __attribute__((ext_vector_type(4)));

#define T_LEN    16384
#define BATCH    4
#define WDC      128          // residual/gate channels
#define SDC      512          // skip channels
#define NLAYER   20
#define FINALMIN 2047         // first valid output timestep (T - finalout)
#define FINALOUT 14337
#define NT       64           // timesteps per workgroup tile

#if __has_builtin(__builtin_amdgcn_global_load_async_to_lds_b128) && \
    __has_builtin(__builtin_amdgcn_s_wait_asynccnt)
#define USE_ASYNC_LDS 1
// Builtin signature (from clang diagnostic): (int4 AS1*, int4 AS3*, imm, imm)
typedef __attribute__((address_space(1))) v4i g4i;
typedef __attribute__((address_space(3))) v4i l4i;
#else
#define USE_ASYNC_LDS 0
#endif

__device__ __forceinline__ v8f vzero() {
    v8f z;
#pragma unroll
    for (int i = 0; i < 8; ++i) z[i] = 0.0f;
    return z;
}

// A-fragment (16x32 f16): lane L<16 holds row M=L, K in kb..kb+15;
// lane L>=16 holds row M=L-16, K in kb+16..kb+31. ld = K-stride of row-major W[M][K].
__device__ __forceinline__ v16h load_a(const _Float16* __restrict__ W, int ld,
                                       int row, int kbase, int hi) {
    const _Float16* p = W + (size_t)row * ld + kbase + hi * 16;
    h8 lo = *(const h8*)p;
    h8 hh = *(const h8*)(p + 8);
    v16h a;
#pragma unroll
    for (int i = 0; i < 8; ++i) { a[i] = lo[i]; a[i + 8] = hh[i]; }
    return a;
}

// B-fragment (32x16 f16): lane L holds column n=L&15, 16 contiguous K values
// starting at kbase + 16*(L>>4). Source is row-major [t][C] f16.
__device__ __forceinline__ v16h load_b16(const _Float16* __restrict__ X, int ld,
                                         int trow, int kbase, int hi) {
    return *(const v16h*)(X + (size_t)trow * ld + kbase + hi * 16);
}

// B-fragment from f32 source with fused ReLU (skip accumulator -> W1 GEMM).
__device__ __forceinline__ v16h load_b_relu_f32(const float* __restrict__ S, int ld,
                                                int trow, int kbase, int hi) {
    const float* p = S + (size_t)trow * ld + kbase + hi * 16;
    v16h b;
#pragma unroll
    for (int i = 0; i < 16; ++i) {
        float v = p[i];
        b[i] = (_Float16)(v > 0.0f ? v : 0.0f);
    }
    return b;
}

__device__ __forceinline__ v8f wmma32(v16h a, v16h b, v8f c) {
    return __builtin_amdgcn_wmma_f32_16x16x32_f16(false, a, false, b, (short)0, c,
                                                  false, false);
}

// ---------------------------------------------------------------------------
// Per-layer fused kernel: gate GEMM (2 taps) -> tanh*sigmoid -> residual GEMM
// (+ shifted x add) -> skip GEMM (+= into f32 skip accumulator).
// Block: 256 threads (8 waves), tile = 64 timesteps, one batch element.
// ---------------------------------------------------------------------------
__global__ __launch_bounds__(256) void layer_kernel(
    const _Float16* __restrict__ xin, _Float16* __restrict__ xout,
    float* __restrict__ skip,
    const _Float16* __restrict__ Wd0, const _Float16* __restrict__ Wd1,
    const float* __restrict__ bd,
    const _Float16* __restrict__ Wr, const float* __restrict__ br,
    const _Float16* __restrict__ Ws, const float* __restrict__ bs,
    int d, int Svalid)
{
    __shared__ _Float16 x0s[NT * WDC];     // x rows [t0-d .. t0-d+63]
    __shared__ _Float16 x1s[NT * WDC];     // x rows [t0   .. t0+63]
    __shared__ _Float16 acts[NT * WDC];    // gated activations, [t][c]
    __shared__ float bds[2 * WDC];
    __shared__ float brs[WDC];
    __shared__ float bss[SDC];

    const int b  = blockIdx.y;
    const int t0 = blockIdx.x * NT;
    if (t0 + NT <= Svalid) return;         // tile fully outside valid region

    const int tid  = threadIdx.x;
    const int w    = tid >> 5;             // wave id 0..7
    const int lane = tid & 31;
    const int nloc = lane & 15;
    const int hi   = lane >> 4;

    const _Float16* xb = xin + (size_t)b * T_LEN * WDC;

    // ---- stage the two x tiles into LDS ------------------------------------
    {
        int r = tid >> 2;                  // row 0..63
        int q = tid & 3;                   // 64B quarter of a 256B row
        int ts0 = t0 - d + r; if (ts0 < 0) ts0 = 0;
        int ts1 = t0 + r;
#if USE_ASYNC_LDS
        // CDNA5 async Global->LDS DMA, bypasses VGPRs; tracked by ASYNCcnt.
        _Float16* s0 = const_cast<_Float16*>(xb + (size_t)ts0 * WDC + q * 32);
        _Float16* s1 = const_cast<_Float16*>(xb + (size_t)ts1 * WDC + q * 32);
        _Float16* d0 = x0s + r * WDC + q * 32;
        _Float16* d1 = x1s + r * WDC + q * 32;
#pragma unroll
        for (int i = 0; i < 4; ++i) {
            __builtin_amdgcn_global_load_async_to_lds_b128(
                (g4i*)(s0 + i * 8), (l4i*)(d0 + i * 8), 0, 0);
            __builtin_amdgcn_global_load_async_to_lds_b128(
                (g4i*)(s1 + i * 8), (l4i*)(d1 + i * 8), 0, 0);
        }
#else
        const uint4* s0 = (const uint4*)(xb + (size_t)ts0 * WDC);
        const uint4* s1 = (const uint4*)(xb + (size_t)ts1 * WDC);
        uint4* d0 = (uint4*)(x0s + r * WDC);
        uint4* d1 = (uint4*)(x1s + r * WDC);
#pragma unroll
        for (int i = 0; i < 4; ++i) {
            d0[q * 4 + i] = s0[q * 4 + i];
            d1[q * 4 + i] = s1[q * 4 + i];
        }
#endif
    }
    for (int i = tid; i < 2 * WDC; i += 256) bds[i] = bd[i];
    for (int i = tid; i < WDC;     i += 256) brs[i] = br[i];
    for (int i = tid; i < SDC;     i += 256) bss[i] = bs[i];
#if USE_ASYNC_LDS
    __builtin_amdgcn_s_wait_asynccnt(0);   // async tile DMA complete
#endif
    __syncthreads();

    // ---- phase 1: in_act = Wd0*x(t-d) + Wd1*x(t); wave w owns M-tiles (w, w+8)
    v8f accT[4], accS[4];
#pragma unroll
    for (int nt = 0; nt < 4; ++nt) { accT[nt] = vzero(); accS[nt] = vzero(); }

    const int rowT = 16 * w + nloc;            // tanh half rows 0..127
    const int rowS = 16 * (w + 8) + nloc;      // sigmoid half rows 128..255
#pragma unroll
    for (int k = 0; k < 4; ++k) {
        const int kb = k * 32;
        v16h aT0 = load_a(Wd0, WDC, rowT, kb, hi);
        v16h aT1 = load_a(Wd1, WDC, rowT, kb, hi);
        v16h aS0 = load_a(Wd0, WDC, rowS, kb, hi);
        v16h aS1 = load_a(Wd1, WDC, rowS, kb, hi);
#pragma unroll
        for (int nt = 0; nt < 4; ++nt) {
            v16h b0 = load_b16(x0s, WDC, nt * 16 + nloc, kb, hi);
            v16h b1 = load_b16(x1s, WDC, nt * 16 + nloc, kb, hi);
            accT[nt] = wmma32(aT0, b0, accT[nt]);
            accT[nt] = wmma32(aT1, b1, accT[nt]);
            accS[nt] = wmma32(aS0, b0, accS[nt]);
            accS[nt] = wmma32(aS1, b1, accS[nt]);
        }
    }

    // ---- phase 2: gated activation -> LDS (B-fragment friendly [t][c] f16) --
    const int cbase = 16 * w + 8 * hi;         // D-tile channel base (M = r + 8*hi)
#pragma unroll
    for (int nt = 0; nt < 4; ++nt) {
        int trow = nt * 16 + nloc;
#pragma unroll
        for (int r = 0; r < 8; r += 2) {
            float tv0 = accT[nt][r]     + bds[cbase + r];
            float tv1 = accT[nt][r + 1] + bds[cbase + r + 1];
            float sv0 = accS[nt][r]     + bds[WDC + cbase + r];
            float sv1 = accS[nt][r + 1] + bds[WDC + cbase + r + 1];
            float a0 = tanhf(tv0) * (1.0f / (1.0f + __expf(-sv0)));
            float a1 = tanhf(tv1) * (1.0f / (1.0f + __expf(-sv1)));
            h2t p; p[0] = (_Float16)a0; p[1] = (_Float16)a1;
            *(h2t*)(acts + trow * WDC + cbase + r) = p;
        }
    }
    __syncthreads();

    // ---- phase 3: residual = Wr*acts + br + x(t) -> xout --------------------
    {
        v8f accR[4];
#pragma unroll
        for (int nt = 0; nt < 4; ++nt) accR[nt] = vzero();
#pragma unroll
        for (int k = 0; k < 4; ++k) {
            v16h aR = load_a(Wr, WDC, rowT, k * 32, hi);
#pragma unroll
            for (int nt = 0; nt < 4; ++nt) {
                v16h bA = load_b16(acts, WDC, nt * 16 + nloc, k * 32, hi);
                accR[nt] = wmma32(aR, bA, accR[nt]);
            }
        }
        _Float16* xo = xout + (size_t)b * T_LEN * WDC;
#pragma unroll
        for (int nt = 0; nt < 4; ++nt) {
            int tloc = nt * 16 + nloc;
            int t = t0 + tloc;
            if (t >= Svalid) {
#pragma unroll
                for (int r = 0; r < 8; r += 2) {
                    int c = cbase + r;
                    float v0 = accR[nt][r]     + brs[c]     + (float)x1s[tloc * WDC + c];
                    float v1 = accR[nt][r + 1] + brs[c + 1] + (float)x1s[tloc * WDC + c + 1];
                    h2t p; p[0] = (_Float16)v0; p[1] = (_Float16)v1;
                    *(h2t*)(xo + (size_t)t * WDC + c) = p;
                }
            }
        }
    }

    // ---- phase 4: skip += Ws*acts + bs (f32 RMW, L2-resident) ---------------
    {
        float* sp = skip + (size_t)b * T_LEN * SDC;
#pragma unroll
        for (int mm = 0; mm < 4; ++mm) {
            int m = 4 * w + mm;                       // wave w owns M-tiles 4w..4w+3
            int rowW = 16 * m + nloc;
            v8f acc[4];
#pragma unroll
            for (int nt = 0; nt < 4; ++nt) acc[nt] = vzero();
#pragma unroll
            for (int k = 0; k < 4; ++k) {
                v16h aS = load_a(Ws, WDC, rowW, k * 32, hi);
#pragma unroll
                for (int nt = 0; nt < 4; ++nt) {
                    v16h bA = load_b16(acts, WDC, nt * 16 + nloc, k * 32, hi);
                    acc[nt] = wmma32(aS, bA, acc[nt]);
                }
            }
            int c0 = 16 * m + 8 * hi;
#pragma unroll
            for (int nt = 0; nt < 4; ++nt) {
                int t = t0 + nt * 16 + nloc;
                if (t >= FINALMIN) {                  // FINALMIN >= Svalid always
                    float4* q = (float4*)(sp + (size_t)t * SDC + c0);
                    float4 v0 = q[0];
                    v0.x += acc[nt][0] + bss[c0 + 0];
                    v0.y += acc[nt][1] + bss[c0 + 1];
                    v0.z += acc[nt][2] + bss[c0 + 2];
                    v0.w += acc[nt][3] + bss[c0 + 3];
                    q[0] = v0;
                    float4 v1 = q[1];
                    v1.x += acc[nt][4] + bss[c0 + 4];
                    v1.y += acc[nt][5] + bss[c0 + 5];
                    v1.z += acc[nt][6] + bss[c0 + 6];
                    v1.w += acc[nt][7] + bss[c0 + 7];
                    q[1] = v1;
                }
            }
        }
    }
}

// ---------------------------------------------------------------------------
// Post kernel 1: h2 = relu(W1 * relu(skip)); M=512, K=512 (f32 B with fused
// relu -> f16 fragments), N=64/block. Wave w owns M-tiles {w, w+8, w+16, w+24}.
// ---------------------------------------------------------------------------
__global__ __launch_bounds__(256) void post1_kernel(
    const float* __restrict__ skip, const _Float16* __restrict__ W1f,
    _Float16* __restrict__ h2)
{
    const int b  = blockIdx.y;
    const int t0 = (blockIdx.x + FINALMIN / NT) * NT;
    const int tid = threadIdx.x;
    const int w = tid >> 5, lane = tid & 31;
    const int nloc = lane & 15, hi = lane >> 4;

    const float* sb = skip + (size_t)b * T_LEN * SDC;
    _Float16*    hb = h2   + (size_t)b * T_LEN * SDC;

    for (int nt = 0; nt < 4; ++nt) {
        int trow = t0 + nt * 16 + nloc;
        v8f acc[4];
#pragma unroll
        for (int mm = 0; mm < 4; ++mm) acc[mm] = vzero();
        for (int k = 0; k < 16; ++k) {
            v16h bfr = load_b_relu_f32(sb, SDC, trow, k * 32, hi);
#pragma unroll
            for (int mm = 0; mm < 4; ++mm) {
                int row = 16 * (w + 8 * mm) + nloc;
                v16h a = load_a(W1f, SDC, row, k * 32, hi);
                acc[mm] = wmma32(a, bfr, acc[mm]);
            }
        }
        int t = t0 + nt * 16 + nloc;
        if (t >= FINALMIN) {
#pragma unroll
            for (int mm = 0; mm < 4; ++mm) {
                int c0 = 16 * (w + 8 * mm) + 8 * hi;
#pragma unroll
                for (int r = 0; r < 8; r += 2) {
                    float v0 = acc[mm][r];     if (v0 < 0.0f) v0 = 0.0f;
                    float v1 = acc[mm][r + 1]; if (v1 < 0.0f) v1 = 0.0f;
                    h2t p; p[0] = (_Float16)v0; p[1] = (_Float16)v1;
                    *(h2t*)(hb + (size_t)t * SDC + c0 + r) = p;
                }
            }
        }
    }
}

// ---------------------------------------------------------------------------
// Post kernel 2: out = W2 * h2; M=256, K=512, output f32 in [B][256][finalout].
// Wave w owns M-tiles {w, w+8}.
// ---------------------------------------------------------------------------
__global__ __launch_bounds__(256) void post2_kernel(
    const _Float16* __restrict__ h2, const _Float16* __restrict__ W2f,
    float* __restrict__ out)
{
    const int b  = blockIdx.y;
    const int t0 = (blockIdx.x + FINALMIN / NT) * NT;
    const int tid = threadIdx.x;
    const int w = tid >> 5, lane = tid & 31;
    const int nloc = lane & 15, hi = lane >> 4;

    const _Float16* hb = h2 + (size_t)b * T_LEN * SDC;

    for (int nt = 0; nt < 4; ++nt) {
        int trow = t0 + nt * 16 + nloc;
        v8f acc0 = vzero(), acc1 = vzero();
        for (int k = 0; k < 16; ++k) {
            v16h bf = load_b16(hb, SDC, trow, k * 32, hi);
            v16h a0 = load_a(W2f, SDC, 16 * w + nloc, k * 32, hi);
            v16h a1 = load_a(W2f, SDC, 16 * (w + 8) + nloc, k * 32, hi);
            acc0 = wmma32(a0, bf, acc0);
            acc1 = wmma32(a1, bf, acc1);
        }
        int t = t0 + nt * 16 + nloc;
        if (t >= FINALMIN) {
            int trel = t - FINALMIN;
#pragma unroll
            for (int r = 0; r < 8; ++r) {
                int o0 = 16 * w + r + 8 * hi;
                out[((size_t)(b * 256 + o0)) * FINALOUT + trel] = acc0[r];
                int o1 = o0 + 128;
                out[((size_t)(b * 256 + o1)) * FINALOUT + trel] = acc1[r];
            }
        }
    }
}

// ---------------------------------------------------------------------------
// Small setup kernels
// ---------------------------------------------------------------------------
__global__ void embed_kernel(const int* __restrict__ y,
                             const float* __restrict__ embed,
                             _Float16* __restrict__ x)
{
    int bt = blockIdx.x;
    int c  = threadIdx.x;
    int v  = y[bt];
    x[(size_t)bt * WDC + c] = (_Float16)embed[(size_t)v * WDC + c];
}

__global__ void cvt_kernel(const float* __restrict__ src,
                           _Float16* __restrict__ dst, int n)
{
    int i = blockIdx.x * 256 + threadIdx.x;
    if (i < n) dst[i] = (_Float16)src[i];
}

// Wd layout [L][256][128][2] -> split into two tap-contiguous f16 matrices.
__global__ void cvt_wd_kernel(const float* __restrict__ Wd,
                              _Float16* __restrict__ wd0,
                              _Float16* __restrict__ wd1, int n)
{
    int i = blockIdx.x * 256 + threadIdx.x;
    if (i < n) {
        wd0[i] = (_Float16)Wd[2 * (size_t)i];
        wd1[i] = (_Float16)Wd[2 * (size_t)i + 1];
    }
}

__global__ void zero_kernel(float* __restrict__ p, long n)
{
    long i = (long)blockIdx.x * 256 + threadIdx.x;
    if (i < n) p[i] = 0.0f;
}

// ---------------------------------------------------------------------------
extern "C" void kernel_launch(void* const* d_in, const int* in_sizes, int n_in,
                              void* d_out, int out_size, void* d_ws, size_t ws_size,
                              hipStream_t stream)
{
    (void)in_sizes; (void)n_in; (void)out_size; (void)ws_size;

    const int*   y     = (const int*)d_in[0];
    const float* embed = (const float*)d_in[1];
    const float* Wd    = (const float*)d_in[2];
    const float* bd    = (const float*)d_in[3];
    const float* Ws    = (const float*)d_in[4];
    const float* bs    = (const float*)d_in[5];
    const float* Wr    = (const float*)d_in[6];
    const float* br    = (const float*)d_in[7];
    const float* W1    = (const float*)d_in[8];
    const float* W2    = (const float*)d_in[9];
    float* out = (float*)d_out;

    char* ws = (char*)d_ws;
    size_t off = 0;
    auto alloc = [&](size_t bytes) {
        size_t o = off;
        off += (bytes + 255) & ~(size_t)255;
        return o;
    };

    _Float16* xA    = (_Float16*)(ws + alloc((size_t)BATCH * T_LEN * WDC * 2));
    _Float16* xB    = (_Float16*)(ws + alloc((size_t)BATCH * T_LEN * WDC * 2));
    float*    skip  = (float*)   (ws + alloc((size_t)BATCH * T_LEN * SDC * 4));
    _Float16* h2buf = (_Float16*)(ws + alloc((size_t)BATCH * T_LEN * SDC * 2));
    _Float16* wd0   = (_Float16*)(ws + alloc((size_t)NLAYER * 2 * WDC * WDC * 2));
    _Float16* wd1   = (_Float16*)(ws + alloc((size_t)NLAYER * 2 * WDC * WDC * 2));
    _Float16* wrf   = (_Float16*)(ws + alloc((size_t)NLAYER * WDC * WDC * 2));
    _Float16* wsf   = (_Float16*)(ws + alloc((size_t)NLAYER * SDC * WDC * 2));
    _Float16* w1f   = (_Float16*)(ws + alloc((size_t)SDC * SDC * 2));
    _Float16* w2f   = (_Float16*)(ws + alloc((size_t)256 * SDC * 2));

    // --- weight conversion f32 -> f16 (deterministic, every call) -----------
    int nWd = NLAYER * 2 * WDC * WDC;
    cvt_wd_kernel<<<(nWd + 255) / 256, 256, 0, stream>>>(Wd, wd0, wd1, nWd);
    int nWr = NLAYER * WDC * WDC;
    cvt_kernel<<<(nWr + 255) / 256, 256, 0, stream>>>(Wr, wrf, nWr);
    int nWs = NLAYER * SDC * WDC;
    cvt_kernel<<<(nWs + 255) / 256, 256, 0, stream>>>(Ws, wsf, nWs);
    cvt_kernel<<<(SDC * SDC + 255) / 256, 256, 0, stream>>>(W1, w1f, SDC * SDC);
    cvt_kernel<<<(256 * SDC + 255) / 256, 256, 0, stream>>>(W2, w2f, 256 * SDC);

    long nSkip = (long)BATCH * T_LEN * SDC;
    zero_kernel<<<(unsigned)((nSkip + 255) / 256), 256, 0, stream>>>(skip, nSkip);

    embed_kernel<<<BATCH * T_LEN, WDC, 0, stream>>>(y, embed, xA);

    // --- 20 fused dilated-conv layers ---------------------------------------
    static const int dil[NLAYER] = {1, 2, 4, 8, 16, 32, 64, 128, 256, 512,
                                    1, 2, 4, 8, 16, 32, 64, 128, 256, 512};
    int S = 0;
    _Float16* xin = xA;
    _Float16* xout = xB;
    for (int l = 0; l < NLAYER; ++l) {
        int d = dil[l];
        int Svalid = S + d;
        layer_kernel<<<dim3(T_LEN / NT, BATCH), 256, 0, stream>>>(
            xin, xout, skip,
            wd0 + (size_t)l * 2 * WDC * WDC, wd1 + (size_t)l * 2 * WDC * WDC,
            bd + (size_t)l * 2 * WDC,
            wrf + (size_t)l * WDC * WDC, br + (size_t)l * WDC,
            wsf + (size_t)l * SDC * WDC, bs + (size_t)l * SDC,
            d, Svalid);
        S += d;
        _Float16* tmp = xin; xin = xout; xout = tmp;
    }

    // --- post 1x1 convs ------------------------------------------------------
    const int firstTile = FINALMIN / NT;                  // tile 31
    const int postTiles = T_LEN / NT - firstTile;         // 225
    post1_kernel<<<dim3(postTiles, BATCH), 256, 0, stream>>>(skip, w1f, h2buf);
    post2_kernel<<<dim3(postTiles, BATCH), 256, 0, stream>>>(h2buf, w2f, out);
}
